// FixedVectorQuantizer_87041807220994
// MI455X (gfx1250) — matrix operand order repaired and checked by reading, not compiled, not run on hardware
//
#include <hip/hip_runtime.h>

// ---------------------------------------------------------------------------
// FixedVectorQuantizer for MI455X (gfx1250, wave32, WMMA).
//   out[0] : quantized = label_mat[argmin_k dist]          [16384 x 512] f32
//   out[1] : -distance = 2*x@c^T - ||x||^2 - ||c||^2       [16384 x 4096] f32
// GEMM in split bf16 (hi+lo residual) via v_wmma_f32_16x16x32_bf16 (fp32-grade
// argmin at bf16 matrix-pipe speed). Fast path pre-splits inputs into bf16
// planes in workspace and stages tiles with GLOBAL_LOAD_ASYNC_TO_LDS_B128
// (ASYNCcnt, LDS double-buffer); fallback fuses the conversion in-kernel.
// ---------------------------------------------------------------------------

#define B_ROWS  16384
#define K_CODES 4096
#define D_DIM   512
#define LDA     40      // bf16 elements per padded LDS row (80B -> bank spread)

typedef __attribute__((ext_vector_type(16))) __bf16 bf16x16;
typedef __attribute__((ext_vector_type(8)))  float  floatx8;

union FragU { uint4 q[2]; bf16x16 v; };

__device__ __forceinline__ unsigned ford(float f) {
    unsigned u = __float_as_uint(f);
    return (u & 0x80000000u) ? ~u : (u | 0x80000000u);
}

__device__ __forceinline__ unsigned long long shflxor64(unsigned long long v, int m) {
    unsigned lo = (unsigned)(v & 0xFFFFFFFFull);
    unsigned hi = (unsigned)(v >> 32);
    lo = __shfl_xor(lo, m, 32);
    hi = __shfl_xor(hi, m, 32);
    return ((unsigned long long)hi << 32) | lo;
}

// f32 -> bf16 hi + bf16 lo (exact residual by Sterbenz; ~16 mantissa bits kept)
__device__ __forceinline__ void splitbf(float f, unsigned short& h, unsigned short& l) {
    unsigned bits = __float_as_uint(f);
    h = (unsigned short)(bits >> 16);
    float r = f - __uint_as_float(bits & 0xFFFF0000u);
    l = (unsigned short)(__float_as_uint(r) >> 16);
}

// --- CDNA5 async global->LDS copy (ASYNCcnt-tracked), 16B per active lane ---
__device__ __forceinline__ void async_cp16(unsigned lds_off, const void* gptr) {
    unsigned long long ga = (unsigned long long)(uintptr_t)gptr;
    asm volatile("global_load_async_to_lds_b128 %0, %1, off"
                 :: "v"(lds_off), "v"(ga) : "memory");
}
__device__ __forceinline__ void wait_async0() {
    asm volatile("s_wait_asynccnt 0x0" ::: "memory");
}
__device__ __forceinline__ unsigned lds_addr_of(const void* p) {
    // generic shared pointer: low 32 bits are the LDS byte offset
    return (unsigned)(uintptr_t)p;
}

// ---------------------------------------------------------------------------
// per-row squared norm (D=512, 128 thr = 1 float4/thread) + argmin-key init
// ---------------------------------------------------------------------------
__global__ __launch_bounds__(128)
void rownorm_init_kernel(const float* __restrict__ src,
                         float* __restrict__ norms,
                         unsigned long long* __restrict__ keys) {
    const int row = blockIdx.x;
    const int t   = threadIdx.x;
    float4 v = ((const float4*)(src + (size_t)row * D_DIM))[t];
    float s = v.x * v.x + v.y * v.y + v.z * v.z + v.w * v.w;
#pragma unroll
    for (int m = 1; m < 32; m <<= 1) s += __shfl_xor(s, m, 32);
    __shared__ float part[4];
    if ((t & 31) == 0) part[t >> 5] = s;
    __syncthreads();
    if (t == 0) {
        norms[row] = part[0] + part[1] + part[2] + part[3];
        if (keys) keys[row] = 0xFFFFFFFFFFFFFFFFull;
    }
}

// ---------------------------------------------------------------------------
// one-shot f32 -> (bf16 hi plane, bf16 lo plane), 4 elements / thread
// ---------------------------------------------------------------------------
__global__ __launch_bounds__(256)
void split_planes_kernel(const float* __restrict__ src,
                         unsigned short* __restrict__ hi,
                         unsigned short* __restrict__ lo) {
    const size_t i = ((size_t)blockIdx.x * 256 + threadIdx.x) * 4;
    float4 f = *(const float4*)(src + i);
    unsigned short h0, l0, h1, l1, h2, l2, h3, l3;
    splitbf(f.x, h0, l0); splitbf(f.y, h1, l1);
    splitbf(f.z, h2, l2); splitbf(f.w, h3, l3);
    *(ushort4*)(hi + i) = make_ushort4(h0, h1, h2, h3);
    *(ushort4*)(lo + i) = make_ushort4(l0, l1, l2, l3);
}

// ---------------------------------------------------------------------------
// shared epilogue: stream -distance, per-row argmin via packed u64 atomicMin
// ---------------------------------------------------------------------------
__device__ __forceinline__ void vq_epilogue(floatx8 (&acc)[2][2],
                                            int rowBlk, int colBlk,
                                            int waveM, int waveN,
                                            int laneLo, int halfS,
                                            const float* __restrict__ xnorm,
                                            const float* __restrict__ cnorm,
                                            unsigned long long* __restrict__ keys,
                                            float* __restrict__ out2) {
    float cnv[2];
    int   colG[2];
#pragma unroll
    for (int ni = 0; ni < 2; ++ni) {
        colG[ni] = colBlk + waveN * 32 + ni * 16 + laneLo;
        cnv[ni]  = cnorm[colG[ni]];
    }
#pragma unroll
    for (int mi = 0; mi < 2; ++mi) {
#pragma unroll
        for (int v = 0; v < 8; ++v) {
            // C/D layout: VGPR v -> M = v (lanes 0-15) / v+8 (lanes 16-31)
            const int   rowG = rowBlk + waveM * 32 + mi * 16 + halfS * 8 + v;
            const float xnv  = xnorm[rowG];
            unsigned long long best = 0xFFFFFFFFFFFFFFFFull;
#pragma unroll
            for (int ni = 0; ni < 2; ++ni) {
                const float dist = xnv + cnv[ni] - 2.0f * acc[mi][ni][v];
                __builtin_nontemporal_store(-dist,
                    &out2[(size_t)rowG * K_CODES + colG[ni]]);
                unsigned long long key =
                    ((unsigned long long)ford(dist) << 32) | (unsigned)colG[ni];
                best = (key < best) ? key : best;
            }
#pragma unroll
            for (int m = 1; m < 16; m <<= 1) {
                unsigned long long o = shflxor64(best, m);
                best = (o < best) ? o : best;
            }
            if (laneLo == 0) atomicMin(&keys[rowG], best);
        }
    }
}

// ---------------------------------------------------------------------------
// fragment loads (ISA layouts).
// A 16x32 bf16: lane<16 holds K {hb..hb+7, 16+hb..16+hb+7}, hb = half*8.
// B 32x16 bf16: same K order as A: {hb..hb+7, 16+hb..16+hb+7}, hb = half*8.
// ---------------------------------------------------------------------------
__device__ __forceinline__ bf16x16 fragA(const unsigned short* base, int r) {
    FragU u;
    u.q[0] = *(const uint4*)&base[r];
    u.q[1] = *(const uint4*)&base[r + 16];
    return u.v;
}
__device__ __forceinline__ bf16x16 fragB(const unsigned short* base, int r) {
    FragU u;
    u.q[0] = *(const uint4*)&base[r];
    u.q[1] = *(const uint4*)&base[r + 16];
    return u.v;
}

__device__ __forceinline__ void do_wmma(floatx8 (&acc)[2][2],
                                        bf16x16 (&ah)[2], bf16x16 (&al)[2],
                                        bf16x16 (&bh)[2], bf16x16 (&bl)[2]) {
#pragma unroll
    for (int mi = 0; mi < 2; ++mi)
#pragma unroll
        for (int ni = 0; ni < 2; ++ni) {
            acc[mi][ni] = __builtin_amdgcn_wmma_f32_16x16x32_bf16(
                false, ah[mi], false, bh[ni], (short)0, acc[mi][ni], false, false);
            acc[mi][ni] = __builtin_amdgcn_wmma_f32_16x16x32_bf16(
                false, ah[mi], false, bl[ni], (short)0, acc[mi][ni], false, false);
            acc[mi][ni] = __builtin_amdgcn_wmma_f32_16x16x32_bf16(
                false, al[mi], false, bh[ni], (short)0, acc[mi][ni], false, false);
        }
}

// ---------------------------------------------------------------------------
// FAST PATH GEMM: pre-split bf16 planes, async-to-LDS staging, double buffer.
// 128(M) x 64(N) block tile, 8 waves (4x2), wave tile 32x32 (2x2 WMMA accs).
// ---------------------------------------------------------------------------
__global__ __launch_bounds__(256)
void vq_dist_async_kernel(const unsigned short* __restrict__ xhi,
                          const unsigned short* __restrict__ xlo,
                          const unsigned short* __restrict__ lhi,
                          const unsigned short* __restrict__ llo,
                          const float* __restrict__ xnorm,
                          const float* __restrict__ cnorm,
                          unsigned long long* __restrict__ keys,
                          float* __restrict__ out2) {
    __shared__ __align__(16) unsigned short sA[2][2][128 * LDA]; // [buf][hi/lo]
    __shared__ __align__(16) unsigned short sB[2][2][64 * LDA];

    const int t      = threadIdx.x;
    const int lane   = t & 31;
    const int wave   = t >> 5;
    const int waveM  = wave >> 1;    // 0..3
    const int waveN  = wave & 1;     // 0..1
    const int laneLo = lane & 15;
    const int halfS  = lane >> 4;

    const int rowBlk = blockIdx.y * 128;
    const int colBlk = blockIdx.x * 64;

    // staging: 16B segments. A: 128 rows x 4 segs per plane (2/thr/plane),
    // B: 64 rows x 4 segs per plane (1/thr/plane). 6 asyncs / thread / chunk.
    const int ar0 = t >> 2,          as0 = t & 3;
    const int ar1 = (t + 256) >> 2,  as1 = (t + 256) & 3;
    const int br  = t >> 2,          bs  = t & 3;

    const size_t gA0 = (size_t)(rowBlk + ar0) * D_DIM + as0 * 8;
    const size_t gA1 = (size_t)(rowBlk + ar1) * D_DIM + as1 * 8;
    const size_t gB  = (size_t)(colBlk + br)  * D_DIM + bs  * 8;

    unsigned lA0[2][2], lA1[2][2], lB[2][2];
#pragma unroll
    for (int b = 0; b < 2; ++b)
#pragma unroll
        for (int p = 0; p < 2; ++p) {
            lA0[b][p] = lds_addr_of(&sA[b][p][ar0 * LDA + as0 * 8]);
            lA1[b][p] = lds_addr_of(&sA[b][p][ar1 * LDA + as1 * 8]);
            lB [b][p] = lds_addr_of(&sB[b][p][br  * LDA + bs  * 8]);
        }

    auto prefetch = [&](int k, int buf) {
        const size_t off = (size_t)k * 32;  // bf16 elements along D
        async_cp16(lA0[buf][0], xhi + gA0 + off);
        async_cp16(lA0[buf][1], xlo + gA0 + off);
        async_cp16(lA1[buf][0], xhi + gA1 + off);
        async_cp16(lA1[buf][1], xlo + gA1 + off);
        async_cp16(lB [buf][0], lhi + gB  + off);
        async_cp16(lB [buf][1], llo + gB  + off);
    };

    floatx8 acc[2][2] = {};
    prefetch(0, 0);

    for (int k = 0; k < D_DIM / 32; ++k) {
        const int cur = k & 1;
        wait_async0();        // my copies for chunk k have landed in LDS
        __syncthreads();      // everyone's copies visible; prev buf free
        if (k + 1 < D_DIM / 32) prefetch(k + 1, cur ^ 1);

        bf16x16 ah[2], al[2], bh[2], bl[2];
#pragma unroll
        for (int mi = 0; mi < 2; ++mi) {
            const int r = (waveM * 32 + mi * 16 + laneLo) * LDA + halfS * 8;
            ah[mi] = fragA(sA[cur][0], r);
            al[mi] = fragA(sA[cur][1], r);
        }
#pragma unroll
        for (int ni = 0; ni < 2; ++ni) {
            const int r = (waveN * 32 + ni * 16 + laneLo) * LDA + halfS * 8;
            bh[ni] = fragB(sB[cur][0], r);
            bl[ni] = fragB(sB[cur][1], r);
        }
        do_wmma(acc, ah, al, bh, bl);
    }

    vq_epilogue(acc, rowBlk, colBlk, waveM, waveN, laneLo, halfS,
                xnorm, cnorm, keys, out2);
}

// ---------------------------------------------------------------------------
// FALLBACK GEMM (workspace too small): fused f32->bf16 split during staging.
// ---------------------------------------------------------------------------
__global__ __launch_bounds__(256)
void vq_dist_fused_kernel(const float* __restrict__ x,
                          const float* __restrict__ label,
                          const float* __restrict__ xnorm,
                          const float* __restrict__ cnorm,
                          unsigned long long* __restrict__ keys,
                          float* __restrict__ out2) {
    __shared__ __align__(16) unsigned short sAhi[128 * LDA];
    __shared__ __align__(16) unsigned short sAlo[128 * LDA];
    __shared__ __align__(16) unsigned short sBhi[64 * LDA];
    __shared__ __align__(16) unsigned short sBlo[64 * LDA];

    const int t      = threadIdx.x;
    const int lane   = t & 31;
    const int wave   = t >> 5;
    const int waveM  = wave >> 1;
    const int waveN  = wave & 1;
    const int laneLo = lane & 15;
    const int halfS  = lane >> 4;

    const int rowBlk = blockIdx.y * 128;
    const int colBlk = blockIdx.x * 64;

    const int aRow = t >> 1, aHalf = t & 1;
    const int bRow = t >> 2, bQ = t & 3;

    floatx8 acc[2][2] = {};

    for (int kc = 0; kc < D_DIM; kc += 32) {
        __syncthreads();
        {
            const float4* s4 =
                (const float4*)(x + (size_t)(rowBlk + aRow) * D_DIM + kc + aHalf * 16);
            unsigned hp[8], lp[8];
#pragma unroll
            for (int i = 0; i < 4; ++i) {
                float4 f = s4[i];
                unsigned short h0, l0, h1, l1, h2, l2, h3, l3;
                splitbf(f.x, h0, l0); splitbf(f.y, h1, l1);
                splitbf(f.z, h2, l2); splitbf(f.w, h3, l3);
                hp[2 * i]     = (unsigned)h0 | ((unsigned)h1 << 16);
                hp[2 * i + 1] = (unsigned)h2 | ((unsigned)h3 << 16);
                lp[2 * i]     = (unsigned)l0 | ((unsigned)l1 << 16);
                lp[2 * i + 1] = (unsigned)l2 | ((unsigned)l3 << 16);
            }
            uint4* dh = (uint4*)&sAhi[aRow * LDA + aHalf * 16];
            uint4* dl = (uint4*)&sAlo[aRow * LDA + aHalf * 16];
            dh[0] = make_uint4(hp[0], hp[1], hp[2], hp[3]);
            dh[1] = make_uint4(hp[4], hp[5], hp[6], hp[7]);
            dl[0] = make_uint4(lp[0], lp[1], lp[2], lp[3]);
            dl[1] = make_uint4(lp[4], lp[5], lp[6], lp[7]);
        }
        {
            const float4* s4 =
                (const float4*)(label + (size_t)(colBlk + bRow) * D_DIM + kc + bQ * 8);
            unsigned hp[4], lp[4];
#pragma unroll
            for (int i = 0; i < 2; ++i) {
                float4 f = s4[i];
                unsigned short h0, l0, h1, l1, h2, l2, h3, l3;
                splitbf(f.x, h0, l0); splitbf(f.y, h1, l1);
                splitbf(f.z, h2, l2); splitbf(f.w, h3, l3);
                hp[2 * i]     = (unsigned)h0 | ((unsigned)h1 << 16);
                hp[2 * i + 1] = (unsigned)h2 | ((unsigned)h3 << 16);
                lp[2 * i]     = (unsigned)l0 | ((unsigned)l1 << 16);
                lp[2 * i + 1] = (unsigned)l2 | ((unsigned)l3 << 16);
            }
            *(uint4*)&sBhi[bRow * LDA + bQ * 8] = make_uint4(hp[0], hp[1], hp[2], hp[3]);
            *(uint4*)&sBlo[bRow * LDA + bQ * 8] = make_uint4(lp[0], lp[1], lp[2], lp[3]);
        }
        __syncthreads();

        bf16x16 ah[2], al[2], bh[2], bl[2];
#pragma unroll
        for (int mi = 0; mi < 2; ++mi) {
            const int r = (waveM * 32 + mi * 16 + laneLo) * LDA + halfS * 8;
            ah[mi] = fragA(sAhi, r);
            al[mi] = fragA(sAlo, r);
        }
#pragma unroll
        for (int ni = 0; ni < 2; ++ni) {
            const int r = (waveN * 32 + ni * 16 + laneLo) * LDA + halfS * 8;
            bh[ni] = fragB(sBhi, r);
            bl[ni] = fragB(sBlo, r);
        }
        do_wmma(acc, ah, al, bh, bl);
    }

    vq_epilogue(acc, rowBlk, colBlk, waveM, waveN, laneLo, halfS,
                xnorm, cnorm, keys, out2);
}

// ---------------------------------------------------------------------------
__global__ __launch_bounds__(128)
void vq_gather_kernel(const unsigned long long* __restrict__ keys,
                      const float* __restrict__ label,
                      float* __restrict__ out1) {
    const int row = blockIdx.x;
    const unsigned idx = (unsigned)(keys[row] & 0xFFFFFFFFull);
    ((float4*)(out1 + (size_t)row * D_DIM))[threadIdx.x] =
        ((const float4*)(label + (size_t)idx * D_DIM))[threadIdx.x];
}

// ---------------------------------------------------------------------------
extern "C" void kernel_launch(void* const* d_in, const int* in_sizes, int n_in,
                              void* d_out, int out_size, void* d_ws, size_t ws_size,
                              hipStream_t stream) {
    const float* x     = (const float*)d_in[0];          // [B, D]
    // d_in[1] (`var`) does not influence either reference output.
    const float* label = (const float*)d_in[2];          // [K, D]

    float* out1 = (float*)d_out;                         // quantized [B, D]
    float* out2 = out1 + (size_t)B_ROWS * D_DIM;         // -distance [B, K]

    // workspace: xnorm | cnorm | keys | (optional) bf16 planes
    float* xn = (float*)d_ws;
    float* cn = xn + B_ROWS;
    unsigned long long* keys =
        (unsigned long long*)((char*)d_ws + (size_t)(B_ROWS + K_CODES) * sizeof(float));
    const size_t base = (size_t)(B_ROWS + K_CODES) * 4 + (size_t)B_ROWS * 8; // 212992
    unsigned short* xhi = (unsigned short*)((char*)d_ws + base);
    unsigned short* xlo = xhi + (size_t)B_ROWS * D_DIM;
    unsigned short* lhi = xlo + (size_t)B_ROWS * D_DIM;
    unsigned short* llo = lhi + (size_t)K_CODES * D_DIM;
    const size_t need = base +
        2 * ((size_t)B_ROWS * D_DIM + (size_t)K_CODES * D_DIM) * sizeof(unsigned short);

    rownorm_init_kernel<<<B_ROWS, 128, 0, stream>>>(x, xn, keys);
    rownorm_init_kernel<<<K_CODES, 128, 0, stream>>>(label, cn, nullptr);

    if (ws_size >= need) {
        split_planes_kernel<<<(B_ROWS * D_DIM) / 1024, 256, 0, stream>>>(x, xhi, xlo);
        split_planes_kernel<<<(K_CODES * D_DIM) / 1024, 256, 0, stream>>>(label, lhi, llo);
        vq_dist_async_kernel<<<dim3(K_CODES / 64, B_ROWS / 128), 256, 0, stream>>>(
            xhi, xlo, lhi, llo, xn, cn, keys, out2);
    } else {
        vq_dist_fused_kernel<<<dim3(K_CODES / 64, B_ROWS / 128), 256, 0, stream>>>(
            x, label, xn, cn, keys, out2);
    }

    vq_gather_kernel<<<B_ROWS, 128, 0, stream>>>(keys, label, out1);

    (void)in_sizes; (void)n_in; (void)out_size; (void)ws_size;
}
